// TransformerBlock_5970004542013
// MI455X (gfx1250) — matrix operand order repaired and checked by reading, not compiled, not run on hardware
//
#include <hip/hip_runtime.h>
#include <hip/hip_bf16.h>
#include <math.h>

// ---------------------------------------------------------------------------
// TransformerBlock for MI455X (gfx1250): bf16 WMMA GEMMs + flash attention.
// B=2, T=2048, C=1024, H=16, HD=64. All matmuls via v_wmma_f32_16x16x32_bf16.
// Global->LDS staging uses the CDNA5 async path (global_load_async_to_lds_b128
// + s_wait_asynccnt); all LDS fragment reads are contiguous 16B per lane.
// ---------------------------------------------------------------------------

typedef __attribute__((ext_vector_type(16))) __bf16 v16bf;
typedef __attribute__((ext_vector_type(8)))  float  v8f;

#define Bc 2
#define Tc 2048
#define Cc 1024
#define Hc 16
#define HDc 64

__device__ __forceinline__ v8f wmma_bf16(v16bf a, v16bf b, v8f c) {
    // (neg_a, A, neg_b, B, c_mod, C, reuse_a, reuse_b)
    return __builtin_amdgcn_wmma_f32_16x16x32_bf16(false, a, false, b,
                                                   (short)0, c, false, false);
}

// --- CDNA5 async global->LDS copy (16 bytes per lane, ASYNCcnt-tracked) -----
__device__ __forceinline__ void async_ld16(const void* gsrc, void* lds_dst) {
    asm volatile("global_load_async_to_lds_b128 %0, %1, off"
                 :: "v"((unsigned)(size_t)lds_dst), "v"(gsrc)
                 : "memory");
}
__device__ __forceinline__ void wait_async() {
    asm volatile("s_wait_asynccnt 0x0" ::: "memory");
}

// --- fragment loaders, per CDNA5 ISA 7.12.2 layouts (wave32) ----------------
// A (16x32 bf16, row-major, leading dim lda): lane m = lane&15; per lane the
// fragment is two contiguous 8-half runs at k = hal*8 and k = 16 + hal*8.
__device__ __forceinline__ v16bf load_A(const __bf16* base, int lda) {
    const int lane = threadIdx.x & 31;
    const __bf16* p = base + (lane & 15) * lda + (lane >> 4) * 8;
    union { uint4 u[2]; v16bf v; } t;
    t.u[0] = *(const uint4*)p;          // k0 .. k0+7
    t.u[1] = *(const uint4*)(p + 16);   // k0+16 .. k0+23
    return t.v;
}

// Same A layout but f32 source (converted to bf16 in flight).
__device__ __forceinline__ v16bf load_A_f32(const float* base, int lda) {
    const int lane = threadIdx.x & 31;
    const float* p = base + (lane & 15) * lda + (lane >> 4) * 8;
    const float4 f0 = *(const float4*)p;
    const float4 f1 = *(const float4*)(p + 4);
    const float4 f2 = *(const float4*)(p + 16);
    const float4 f3 = *(const float4*)(p + 20);
    v16bf a;
    a[0] = (__bf16)f0.x;  a[1] = (__bf16)f0.y;
    a[2] = (__bf16)f0.z;  a[3] = (__bf16)f0.w;
    a[4] = (__bf16)f1.x;  a[5] = (__bf16)f1.y;
    a[6] = (__bf16)f1.z;  a[7] = (__bf16)f1.w;
    a[8] = (__bf16)f2.x;  a[9] = (__bf16)f2.y;
    a[10] = (__bf16)f2.z; a[11] = (__bf16)f2.w;
    a[12] = (__bf16)f3.x; a[13] = (__bf16)f3.y;
    a[14] = (__bf16)f3.z; a[15] = (__bf16)f3.w;
    return a;
}

// B (32x16 bf16) from an [n][k] (transposed) tile: element (k,n) = base[n*ld+k].
// lane n = lane&15; per lane two contiguous 8-half runs at k = hal*8 and
// k = 16 + hal*8 (2 x b128), matching the A layout.
__device__ __forceinline__ v16bf load_Bt(const __bf16* base, int ld) {
    const int lane = threadIdx.x & 31;
    const __bf16* p = base + (lane & 15) * ld + (lane >> 4) * 8;
    union { uint4 u[2]; v16bf v; } t;
    t.u[0] = *(const uint4*)p;
    t.u[1] = *(const uint4*)(p + 16);
    return t.v;
}

// ---------------------------------------------------------------------------
// fp32 [K][N] -> bf16 transposed [N][K]  (weights, once per launch)
// ---------------------------------------------------------------------------
__global__ void transpose_to_bf16_kernel(const float* __restrict__ in,
                                         __bf16* __restrict__ out,
                                         int K, int N) {
    int i = blockIdx.x * blockDim.x + threadIdx.x;
    const int stride = gridDim.x * blockDim.x;
    const int total = K * N;
    for (; i < total; i += stride) {
        const int n = i / K;
        const int k = i - n * K;
        out[i] = (__bf16)in[(size_t)k * N + n];
    }
}

// ---------------------------------------------------------------------------
// LayerNorm (one block per row, C=1024) -> bf16
// ---------------------------------------------------------------------------
__global__ __launch_bounds__(256)
void ln_bf16_kernel(const float* __restrict__ x, const float* __restrict__ w,
                    __bf16* __restrict__ out, int Cdim) {
    const int row = blockIdx.x;
    const float* xr = x + (size_t)row * Cdim;
    float s = 0.f, s2 = 0.f;
    for (int c = threadIdx.x; c < Cdim; c += blockDim.x) {
        float v = xr[c];
        s += v; s2 += v * v;
    }
#pragma unroll
    for (int off = 16; off; off >>= 1) {
        s  += __shfl_xor(s, off);
        s2 += __shfl_xor(s2, off);
    }
    __shared__ float sh[16];
    const int wid = threadIdx.x >> 5;
    if ((threadIdx.x & 31) == 0) { sh[wid] = s; sh[8 + wid] = s2; }
    __syncthreads();
    s = 0.f; s2 = 0.f;
#pragma unroll
    for (int i = 0; i < 8; ++i) { s += sh[i]; s2 += sh[8 + i]; }
    const float mu = s / Cdim;
    const float var = s2 / Cdim - mu * mu;
    const float rs = rsqrtf(var + 1e-5f);
    for (int c = threadIdx.x; c < Cdim; c += blockDim.x)
        out[(size_t)row * Cdim + c] = (__bf16)((xr[c] - mu) * rs * w[c]);
}

// ---------------------------------------------------------------------------
// Tiled WMMA GEMM: C[M,N] = A[M,K] @ B[K,N], A bf16 row-major [M][K],
// B supplied TRANSPOSED bf16 [N][K]. Block tile 128x64, BK=32, 8 waves (4x2),
// 32x32 per wave. Tiles staged via async global->LDS.
// EPI: 0 = f32 store; 1 = f32 resid+acc; 2 = bf16 gelu(acc+bias);
//      3 = f32 resid+acc+bias.   (M%128==0, N%64==0, K%32==0 here.)
// ---------------------------------------------------------------------------
#define GBM 128
#define GBN 64
#define GBK 32

template <int EPI>
__global__ __launch_bounds__(256)
void gemm_wmma_bf16(const __bf16* __restrict__ A, const __bf16* __restrict__ Bt,
                    int M, int N, int K,
                    const float* __restrict__ bias,
                    const float* __restrict__ resid,
                    float* __restrict__ outF, __bf16* __restrict__ outB) {
    __shared__ __align__(16) __bf16 As[GBM][GBK + 8];   // [m][k]
    __shared__ __align__(16) __bf16 Bs[GBN][GBK + 8];   // [n][k]

    const int tid = threadIdx.x;
    const int lane = tid & 31;
    const int w = tid >> 5;
    const int wr = w >> 1;      // 0..3
    const int wc = w & 1;       // 0..1
    const int m0 = blockIdx.y * GBM;
    const int n0 = blockIdx.x * GBN;

    v8f acc[2][2] = {};

    for (int k0 = 0; k0 < K; k0 += GBK) {
        // stage A tile 128x32 (two async 16B per thread)
#pragma unroll
        for (int rep = 0; rep < 2; ++rep) {
            const int e = rep * 2048 + tid * 8;
            const int r = e >> 5, c = e & 31;
            async_ld16(A + (size_t)(m0 + r) * K + k0 + c, &As[r][c]);
        }
        // stage B^T tile 64x32 (one async 16B per thread)
        {
            const int e = tid * 8;
            const int r = e >> 5, c = e & 31;
            async_ld16(Bt + (size_t)(n0 + r) * K + k0 + c, &Bs[r][c]);
        }
        wait_async();
        __syncthreads();

        const v16bf a0 = load_A(&As[wr * 32][0], GBK + 8);
        const v16bf a1 = load_A(&As[wr * 32 + 16][0], GBK + 8);
        const v16bf b0 = load_Bt(&Bs[wc * 32][0], GBK + 8);
        const v16bf b1 = load_Bt(&Bs[wc * 32 + 16][0], GBK + 8);
        acc[0][0] = wmma_bf16(a0, b0, acc[0][0]);
        acc[0][1] = wmma_bf16(a0, b1, acc[0][1]);
        acc[1][0] = wmma_bf16(a1, b0, acc[1][0]);
        acc[1][1] = wmma_bf16(a1, b1, acc[1][1]);
        __syncthreads();
    }

    // epilogue (C/D layout: n = lane&15, m = VGPRidx + (lane>=16 ? 8 : 0))
    const int hal = lane >> 4;
    const int nn = lane & 15;
#pragma unroll
    for (int mi = 0; mi < 2; ++mi) {
#pragma unroll
        for (int ni = 0; ni < 2; ++ni) {
            const int col = n0 + wc * 32 + ni * 16 + nn;
#pragma unroll
            for (int i = 0; i < 8; ++i) {
                const int row = m0 + wr * 32 + mi * 16 + hal * 8 + i;
                const size_t idx = (size_t)row * N + col;
                float v = acc[mi][ni][i];
                if (EPI == 0) {
                    outF[idx] = v;
                } else if (EPI == 1) {
                    outF[idx] = resid[idx] + v;
                } else if (EPI == 2) {
                    v += bias[col];
                    v = 0.5f * v * (1.0f + erff(v * 0.70710678118654752f));
                    outB[idx] = (__bf16)v;
                } else {
                    v += bias[col];
                    outF[idx] = resid[idx] + v;
                }
            }
        }
    }
}

// ---------------------------------------------------------------------------
// RoPE + qkv split: qkv f32 [B*T, 3C] ->
//   Q (scaled by 1/sqrt(HD)) bf16 [B,H,T,HD]
//   K bf16 [B,H,T,HD]
//   V bf16 TRANSPOSED [B,H,HD,T]   (so P@V B-fragments read contiguous k)
// ---------------------------------------------------------------------------
__global__ __launch_bounds__(256)
void rope_split_kernel(const float* __restrict__ qkv, __bf16* __restrict__ Qo,
                       __bf16* __restrict__ Ko, __bf16* __restrict__ Vt) {
    const int idx = blockIdx.x * blockDim.x + threadIdx.x;
    const int total = Bc * Tc * Hc * (HDc / 2);
    if (idx >= total) return;
    const int d = idx & 31;
    const int h = (idx >> 5) & (Hc - 1);
    const int bt = idx >> 9;
    const int t = bt & (Tc - 1);
    const int b = bt / Tc;
    const int bh = b * Hc + h;

    const float* row = qkv + (size_t)bt * (3 * Cc);
    const float fr = __powf(10000.0f, -(float)(2 * d) * (1.0f / (float)HDc));
    const float ang = (float)t * fr;
    float sn, cs;
    __sincosf(ang, &sn, &cs);

    const float q1 = row[h * HDc + d],          q2 = row[h * HDc + d + 32];
    const float k1 = row[Cc + h * HDc + d],     k2 = row[Cc + h * HDc + d + 32];
    const float v1 = row[2 * Cc + h * HDc + d], v2 = row[2 * Cc + h * HDc + d + 32];

    const size_t ob = ((size_t)bh * Tc + t) * HDc + d;
    const float sc = 0.125f;  // 1/sqrt(HD), folded into Q
    Qo[ob]      = (__bf16)((q1 * cs - q2 * sn) * sc);
    Qo[ob + 32] = (__bf16)((q2 * cs + q1 * sn) * sc);
    Ko[ob]      = (__bf16)(k1 * cs - k2 * sn);
    Ko[ob + 32] = (__bf16)(k2 * cs + k1 * sn);
    const size_t vb = ((size_t)bh * HDc + d) * Tc + t;
    Vt[vb]            = (__bf16)v1;
    Vt[vb + 32 * Tc]  = (__bf16)v2;
}

// ---------------------------------------------------------------------------
// Flash attention (causal). Grid: (T/128, B*H), 256 threads = 8 waves.
// Each wave owns 16 query rows; K/V streamed in 32-kv tiles through LDS via
// async copies. S = Q K^T and O += P V via WMMA; online softmax in registers.
// Output Y bf16 in [B*T, C] layout (row = b*T+q, col = h*64+hd).
// ---------------------------------------------------------------------------
__global__ __launch_bounds__(256)
void flash_attn_kernel(const __bf16* __restrict__ Q, const __bf16* __restrict__ Kg,
                       const __bf16* __restrict__ Vt, __bf16* __restrict__ Y) {
    __shared__ __align__(16) __bf16 Ks[32][72];   // [kv][hd]
    __shared__ __align__(16) __bf16 Vs[64][40];   // [hd][kv]  (transposed tile)
    __shared__ __align__(16) float  Ps[8][16][36];

    const int tid = threadIdx.x;
    const int lane = tid & 31;
    const int w = tid >> 5;
    const int hal = lane >> 4;
    const int nn = lane & 15;

    const int bh = blockIdx.y;            // b*H + h
    const int q0 = blockIdx.x * 128 + w * 16;
    const size_t baseQ = (size_t)bh * Tc * HDc;
    const size_t baseV = (size_t)bh * HDc * Tc;

    const v16bf qf0 = load_A(Q + baseQ + (size_t)q0 * HDc, HDc);
    const v16bf qf1 = load_A(Q + baseQ + (size_t)q0 * HDc + 32, HDc);

    v8f o[4] = {};
    float mi[8], li[8];
#pragma unroll
    for (int i = 0; i < 8; ++i) { mi[i] = -3.0e38f; li[i] = 0.f; }

    const int nkb = blockIdx.x * 4 + 4;   // causal: kv <= q-block end
    for (int kb = 0; kb < nkb; ++kb) {
        const int kv0 = kb * 32;
        {   // K tile: 32 kv rows x 64 hd  (row-major from Kg)
            const int e = tid * 8;
            const int r = e >> 6, c = e & 63;
            async_ld16(Kg + baseQ + (size_t)(kv0 + r) * HDc + c, &Ks[r][c]);
        }
        {   // V tile: 64 hd rows x 32 kv  (rows of transposed Vt)
            const int e = tid * 8;
            const int r = e >> 5, c = e & 31;
            async_ld16(Vt + baseV + (size_t)r * Tc + kv0 + c, &Vs[r][c]);
        }
        wait_async();
        __syncthreads();

        // S tiles: element (k=hd, n=kv) = Ks[n][k] -> contiguous-k loads
        v8f s0 = {}, s1 = {};
        s0 = wmma_bf16(qf0, load_Bt(&Ks[0][0], 72), s0);
        s0 = wmma_bf16(qf1, load_Bt(&Ks[0][32], 72), s0);
        s1 = wmma_bf16(qf0, load_Bt(&Ks[16][0], 72), s1);
        s1 = wmma_bf16(qf1, load_Bt(&Ks[16][32], 72), s1);

        // causal mask
#pragma unroll
        for (int i = 0; i < 8; ++i) {
            const int qrow = q0 + hal * 8 + i;
            if (kv0 + nn > qrow)      s0[i] = -3.0e38f;
            if (kv0 + 16 + nn > qrow) s1[i] = -3.0e38f;
        }

        // online softmax per row (rows live on 16-lane groups; xor 1/2/4/8
        // reductions stay within the group on wave32)
#pragma unroll
        for (int i = 0; i < 8; ++i) {
            float r = fmaxf(s0[i], s1[i]);
            r = fmaxf(r, __shfl_xor(r, 1));
            r = fmaxf(r, __shfl_xor(r, 2));
            r = fmaxf(r, __shfl_xor(r, 4));
            r = fmaxf(r, __shfl_xor(r, 8));
            const float nm = fmaxf(mi[i], r);
            const float alpha = __expf(mi[i] - nm);
            const float p0 = __expf(s0[i] - nm);
            const float p1 = __expf(s1[i] - nm);
            float rs = p0 + p1;
            rs += __shfl_xor(rs, 1);
            rs += __shfl_xor(rs, 2);
            rs += __shfl_xor(rs, 4);
            rs += __shfl_xor(rs, 8);
            li[i] = li[i] * alpha + rs;
            mi[i] = nm;
            o[0][i] *= alpha; o[1][i] *= alpha;
            o[2][i] *= alpha; o[3][i] *= alpha;
            Ps[w][hal * 8 + i][nn]      = p0;
            Ps[w][hal * 8 + i][16 + nn] = p1;
        }

        // O += P @ V : element (k=kv, n=hd) = Vs[n][k] -> contiguous-k loads
        const v16bf pa = load_A_f32(&Ps[w][0][0], 36);
        o[0] = wmma_bf16(pa, load_Bt(&Vs[0][0], 40),  o[0]);
        o[1] = wmma_bf16(pa, load_Bt(&Vs[16][0], 40), o[1]);
        o[2] = wmma_bf16(pa, load_Bt(&Vs[32][0], 40), o[2]);
        o[3] = wmma_bf16(pa, load_Bt(&Vs[48][0], 40), o[3]);
        __syncthreads();
    }

    // normalize and store Y in [B*T, C] (ready to be cproj's A operand)
    const int b = bh / Hc, h = bh % Hc;
#pragma unroll
    for (int i = 0; i < 8; ++i) {
        const float inv = 1.0f / li[i];
        const int qrow = q0 + hal * 8 + i;
        const size_t orow = ((size_t)b * Tc + qrow) * Cc + h * HDc;
        Y[orow + 0 * 16 + nn] = (__bf16)(o[0][i] * inv);
        Y[orow + 1 * 16 + nn] = (__bf16)(o[1][i] * inv);
        Y[orow + 2 * 16 + nn] = (__bf16)(o[2][i] * inv);
        Y[orow + 3 * 16 + nn] = (__bf16)(o[3][i] * inv);
    }
}

// ---------------------------------------------------------------------------
// Host-side launch
// ---------------------------------------------------------------------------
extern "C" void kernel_launch(void* const* d_in, const int* in_sizes, int n_in,
                              void* d_out, int out_size, void* d_ws, size_t ws_size,
                              hipStream_t stream) {
    const float* x       = (const float*)d_in[0];
    const float* ln1_w   = (const float*)d_in[1];
    const float* w_attn  = (const float*)d_in[2];
    const float* w_cproj = (const float*)d_in[3];
    const float* ln2_w   = (const float*)d_in[4];
    const float* w_fc    = (const float*)d_in[5];
    const float* b_fc    = (const float*)d_in[6];
    const float* w_mproj = (const float*)d_in[7];
    const float* b_mproj = (const float*)d_in[8];
    float* out = (float*)d_out;

    const size_t BT = (size_t)Bc * Tc;  // 4096

    // carve workspace (~177 MB)
    char* p = (char*)d_ws;
    auto alloc = [&](size_t bytes) -> void* {
        void* r = (void*)p;
        p += (bytes + 255) & ~(size_t)255;
        return r;
    };
    __bf16* wt_attn  = (__bf16*)alloc((size_t)Cc * 3 * Cc * 2);   // [3C][C]
    __bf16* wt_cproj = (__bf16*)alloc((size_t)Cc * Cc * 2);       // [C][C]
    __bf16* wt_fc    = (__bf16*)alloc((size_t)Cc * 4 * Cc * 2);   // [4C][C]
    __bf16* wt_mproj = (__bf16*)alloc((size_t)4 * Cc * Cc * 2);   // [C][4C]
    __bf16* h1       = (__bf16*)alloc(BT * Cc * 2);
    float*  qkv      = (float*) alloc(BT * 3 * Cc * 4);
    __bf16* Qb       = (__bf16*)alloc(BT * Cc * 2);
    __bf16* Kb       = (__bf16*)alloc(BT * Cc * 2);
    __bf16* Vtb      = (__bf16*)alloc(BT * Cc * 2);
    __bf16* Yb       = (__bf16*)alloc(BT * Cc * 2);
    float*  x2       = (float*) alloc(BT * Cc * 4);
    __bf16* h2       = (__bf16*)alloc(BT * Cc * 2);
    __bf16* fcact    = (__bf16*)alloc(BT * 4 * Cc * 2);
    if ((size_t)(p - (char*)d_ws) > ws_size) return;  // insufficient scratch

    // weights -> bf16, transposed to [N][K]
    transpose_to_bf16_kernel<<<1024, 256, 0, stream>>>(w_attn,  wt_attn,  Cc, 3 * Cc);
    transpose_to_bf16_kernel<<<1024, 256, 0, stream>>>(w_cproj, wt_cproj, Cc, Cc);
    transpose_to_bf16_kernel<<<1024, 256, 0, stream>>>(w_fc,    wt_fc,    Cc, 4 * Cc);
    transpose_to_bf16_kernel<<<1024, 256, 0, stream>>>(w_mproj, wt_mproj, 4 * Cc, Cc);

    // ln1 -> h1 (bf16)
    ln_bf16_kernel<<<(int)BT, 256, 0, stream>>>(x, ln1_w, h1, Cc);

    // qkv = h1 @ w_attn  [4096 x 3072]
    gemm_wmma_bf16<0><<<dim3(3 * Cc / GBN, BT / GBM), 256, 0, stream>>>(
        h1, wt_attn, (int)BT, 3 * Cc, Cc, nullptr, nullptr, qkv, nullptr);

    // RoPE + split -> Q(scaled)/K bf16 [B,H,T,HD], V bf16 [B,H,HD,T]
    rope_split_kernel<<<(Bc * Tc * Hc * (HDc / 2)) / 256, 256, 0, stream>>>(
        qkv, Qb, Kb, Vtb);

    // flash attention -> Yb [B*T, C] bf16
    flash_attn_kernel<<<dim3(Tc / 128, Bc * Hc), 256, 0, stream>>>(Qb, Kb, Vtb, Yb);

    // x2 = x + Yb @ w_cproj
    gemm_wmma_bf16<1><<<dim3(Cc / GBN, BT / GBM), 256, 0, stream>>>(
        Yb, wt_cproj, (int)BT, Cc, Cc, nullptr, x, x2, nullptr);

    // ln2 -> h2 (bf16)
    ln_bf16_kernel<<<(int)BT, 256, 0, stream>>>(x2, ln2_w, h2, Cc);

    // fcact = gelu(h2 @ w_fc + b_fc)  (bf16)
    gemm_wmma_bf16<2><<<dim3(4 * Cc / GBN, BT / GBM), 256, 0, stream>>>(
        h2, wt_fc, (int)BT, 4 * Cc, Cc, b_fc, nullptr, nullptr, fcact);

    // out = x2 + fcact @ w_mproj + b_mproj
    gemm_wmma_bf16<3><<<dim3(Cc / GBN, BT / GBM), 256, 0, stream>>>(
        fcact, wt_mproj, (int)BT, Cc, 4 * Cc, b_mproj, x2, out, nullptr);
}
